// Attention_79224966742132
// MI455X (gfx1250) — matrix operand order repaired and checked by reading, not compiled, not run on hardware
//
#include <hip/hip_runtime.h>
#include <hip/hip_bf16.h>
#include <math.h>

// ---------------------------------------------------------------------------
// Attention block (QKV proj + axial RoPE + flash attention + out proj)
// MI455X / gfx1250. All GEMM stages on v_wmma_f32_16x16x32_f16 with 128-bit
// operand loads; attention K/V tiles double-buffered into LDS via
// global_load_async_to_lds_b128 (ASYNCcnt) and shared by all 8 waves.
// Shapes hardcoded from reference: B=2, N=2048 (8*16*16), C=1152, nh=16,
// hd=72 (padded to 96 = 3 WMMA K-chunks), rope dim 48.
// ---------------------------------------------------------------------------

typedef __attribute__((ext_vector_type(16))) _Float16 v16h;
typedef __attribute__((ext_vector_type(8)))  _Float16 v8h;
typedef __attribute__((ext_vector_type(4)))  _Float16 v4h;
typedef __attribute__((ext_vector_type(8)))  float    v8f;

#define NH     16
#define HD     72
#define HDP    96
#define CIN    1152
#define C3     3456
#define NTOK   2048
#define BATCH  2
#define ROPE_D 48

static __device__ __forceinline__ v8f wmma16(v16h a, v16h b, v8f c) {
  return __builtin_amdgcn_wmma_f32_16x16x32_f16(false, a, false, b,
                                                (short)0, c, false, false);
}

// two 16B-aligned 8xf16 loads -> one WMMA operand register block
static __device__ __forceinline__ v16h ld16(const _Float16* p0,
                                            const _Float16* p1) {
  v8h lo = *(const v8h*)p0;
  v8h hi = *(const v8h*)p1;
  return __builtin_shufflevector(lo, hi, 0, 1, 2, 3, 4, 5, 6, 7, 8, 9, 10, 11,
                                 12, 13, 14, 15);
}

// async 16B global -> LDS copy (per-lane addresses), tracked by ASYNCcnt
static __device__ __forceinline__ void async_cp16(uint32_t lds_off,
                                                  const _Float16* gsrc) {
  uint64_t ga = (uint64_t)(uintptr_t)gsrc;
  asm volatile("global_load_async_to_lds_b128 %0, %1, off"
               :
               : "v"(lds_off), "v"(ga)
               : "memory");
}
static __device__ __forceinline__ void wait_async0() {
  asm volatile("s_wait_asynccnt 0" ::: "memory");
}

// axial rope frequency for flattened position n (t*256+h*16+w), dim d<48
static __device__ __forceinline__ float rope_freq(int n, int d) {
  const float PI = 3.14159265358979323846f;
  int axis = d >> 4;
  int jj   = (d & 15) >> 1;
  float base = (1.0f + 127.0f * (float)jj * (1.0f / 7.0f)) * PI;
  float pos;
  if (axis == 0)      pos = -1.0f + 2.0f * (float)(n >> 8)        * (1.0f / 7.0f);
  else if (axis == 1) pos = -1.0f + 2.0f * (float)((n >> 4) & 15) * (1.0f / 15.0f);
  else                pos = -1.0f + 2.0f * (float)(n & 15)        * (1.0f / 15.0f);
  return pos * base;
}

// ---------------------------------------------------------------------------
// Kernel 0: zero f16 q/k/vt workspace (provides hd 72->96 zero padding)
// ---------------------------------------------------------------------------
__global__ __launch_bounds__(256) void zero_ws_kernel(uint4* p, long n4) {
  long i = (long)blockIdx.x * blockDim.x + threadIdx.x;
  if (i < n4) { uint4 z; z.x = 0; z.y = 0; z.z = 0; z.w = 0; p[i] = z; }
}

// ---------------------------------------------------------------------------
// Kernel 0b: elementwise f32 -> f16 (for x)
// ---------------------------------------------------------------------------
__global__ __launch_bounds__(256) void cvt_f16_kernel(
    const float* __restrict__ in, _Float16* __restrict__ out, long n) {
  long i = ((long)blockIdx.x * blockDim.x + threadIdx.x) * 4;
  if (i < n) {
    float4 f = *(const float4*)(in + i);
    v4h o;
    o[0] = (_Float16)f.x; o[1] = (_Float16)f.y;
    o[2] = (_Float16)f.z; o[3] = (_Float16)f.w;
    *(v4h*)(out + i) = o;
  }
}

// ---------------------------------------------------------------------------
// Kernel 0c: tiled transpose + f32->f16: out[c*R + r] = in[r*Ccols + c]
// ---------------------------------------------------------------------------
__global__ __launch_bounds__(256) void transpose_cvt_kernel(
    const float* __restrict__ in, _Float16* __restrict__ out, int R,
    int Ccols) {
  __shared__ float tile[32][33];
  const int tx = threadIdx.x & 31, ty = threadIdx.x >> 5;  // 32x8
  const int c0 = blockIdx.x * 32, r0 = blockIdx.y * 32;
#pragma unroll
  for (int i = 0; i < 32; i += 8)
    tile[ty + i][tx] = in[(size_t)(r0 + ty + i) * Ccols + c0 + tx];
  __syncthreads();
#pragma unroll
  for (int i = 0; i < 32; i += 8)
    out[(size_t)(c0 + ty + i) * R + r0 + tx] = (_Float16)tile[tx][ty + i];
}

// ---------------------------------------------------------------------------
// Kernel 1: qkv = xh @ WqkvT, fused RoPE, scatter q/k -> [bh][n][HDP],
// v -> transposed [bh][HDP][n]. Wave computes a 16x64 tile (4 wmma / K-step).
// ---------------------------------------------------------------------------
__global__ __launch_bounds__(256) void qkv_rope_kernel(
    const _Float16* __restrict__ xh, const _Float16* __restrict__ wqt,
    _Float16* __restrict__ qh, _Float16* __restrict__ kh,
    _Float16* __restrict__ vt) {
  const int lane = threadIdx.x & 31;
  const int wave = threadIdx.x >> 5;
  const int NB = C3 / 64;                       // 54 col blocks
  const int bm = blockIdx.x / NB, bn = blockIdx.x % NB;
  const int g = lane >> 4, lr = lane & 15;

  const _Float16* arow =
      xh + (size_t)(bm * 128 + wave * 16 + lr) * CIN;
  v8f acc[4];
  v8f z = {};
#pragma unroll
  for (int t = 0; t < 4; ++t) acc[t] = z;

  for (int kk = 0; kk < CIN; kk += 32) {
    v16h a = ld16(arow + kk + g * 16, arow + kk + g * 16 + 8);
#pragma unroll
    for (int t = 0; t < 4; ++t) {
      const _Float16* bp =
          wqt + (size_t)(bn * 64 + t * 16 + lr) * CIN + kk + g * 16;
      acc[t] = wmma16(a, ld16(bp, bp + 8), acc[t]);
    }
  }

#pragma unroll
  for (int t = 0; t < 4; ++t) {
    const int j   = bn * 64 + t * 16 + lr;      // column in [0, 3456)
    const int s   = j / CIN;                    // 0=q 1=k 2=v
    const int rem = j - s * CIN;
    const int hh  = rem / HD;
    const int d   = rem - hh * HD;
#pragma unroll
    for (int r = 0; r < 8; ++r) {
      const int mrow = bm * 128 + wave * 16 + r + 8 * g;
      const int bb = mrow / NTOK, n = mrow - bb * NTOK;
      float v = acc[t][r];
      float partner = __shfl_xor(v, 1, 32);     // rotate-half partner (d^1)
      float outv = v;
      if (s < 2 && d < ROPE_D) {
        float f = rope_freq(n, d);
        float cs = __cosf(f), sn = __sinf(f);
        outv = (d & 1) ? (v * cs + partner * sn) : (v * cs - partner * sn);
      }
      const int bh = bb * NH + hh;
      if (s == 0)
        qh[((size_t)bh * NTOK + n) * HDP + d] = (_Float16)outv;
      else if (s == 1)
        kh[((size_t)bh * NTOK + n) * HDP + d] = (_Float16)outv;
      else
        vt[((size_t)bh * HDP + d) * NTOK + n] = (_Float16)outv;
    }
  }
}

// ---------------------------------------------------------------------------
// Kernel 2: block-cooperative flash attention. All 8 waves of a block share
// one (b,h); K (32x96) and V^T (96x32) tiles are double-buffered into LDS
// with global_load_async_to_lds_b128 and consumed via ds_load_b128.
// Per key tile: 6 wmma for S, 6 wmma for O per wave.
// ---------------------------------------------------------------------------
__global__ __launch_bounds__(256) void attn_kernel(
    const _Float16* __restrict__ qh, const _Float16* __restrict__ kh,
    const _Float16* __restrict__ vt, _Float16* __restrict__ oh) {
  __shared__ __align__(16) _Float16 ksh[2][32 * HDP];   // 2 x 6 KB
  __shared__ __align__(16) _Float16 vsh[2][HDP * 32];   // 2 x 6 KB
  __shared__ __align__(16) _Float16 pbuf[8][16 * 32];   // 8 KB
  const int tid  = threadIdx.x;
  const int lane = tid & 31;
  const int wave = tid >> 5;
  const int bh   = blockIdx.x >> 4;             // b*NH + h in [0,32)
  const int qt   = (blockIdx.x & 15) * 8 + wave;  // 128 q tiles per bh
  const int g = lane >> 4, lr = lane & 15;
  const float scale = 0.11785113019775792f;     // 1/sqrt(72)

  const _Float16* kbase = kh + (size_t)bh * NTOK * HDP;
  const _Float16* vbase = vt + (size_t)bh * HDP * NTOK;

  // stage one 32-key tile (K contiguous 6144B, V^T 96 rows x 64B) into LDS
  auto stage = [&](int buf, int kt) {
    const _Float16* kg = kbase + (size_t)kt * HDP;
    uint32_t koff = (uint32_t)(uintptr_t)&ksh[buf][0];
    for (int i = tid; i < 384; i += 256)
      async_cp16(koff + (uint32_t)i * 16, kg + i * 8);
    uint32_t voff = (uint32_t)(uintptr_t)&vsh[buf][0];
    for (int i = tid; i < 384; i += 256) {
      int row = i >> 2, seg = i & 3;
      async_cp16(voff + (uint32_t)i * 16,
                 vbase + (size_t)row * NTOK + kt + seg * 8);
    }
  };

  // Q tile in A-layout registers, attention scale folded in.
  const _Float16* qbase = qh + ((size_t)bh * NTOK + qt * 16 + lr) * HDP;
  v16h aq[3];
#pragma unroll
  for (int c = 0; c < 3; ++c) {
    v16h q = ld16(qbase + c * 32 + g * 16, qbase + c * 32 + g * 16 + 8);
#pragma unroll
    for (int i = 0; i < 16; ++i)
      aq[c][i] = (_Float16)((float)q[i] * scale);
  }

  v8f zero8 = {};
  v8f oacc[6];
#pragma unroll
  for (int j = 0; j < 6; ++j) oacc[j] = zero8;
  float mx[8], sm[8];
#pragma unroll
  for (int r = 0; r < 8; ++r) { mx[r] = -1e30f; sm[r] = 0.0f; }

  _Float16* myp = &pbuf[wave][0];

  stage(0, 0);
  int buf = 0;
  for (int kt = 0; kt < NTOK; kt += 32) {
    wait_async0();          // my async copies into `buf` have landed
    __syncthreads();        // everyone's copies have landed
    if (kt + 32 < NTOK) stage(buf ^ 1, kt + 32);   // overlap next tile

    const _Float16* kl = &ksh[buf][0];
    const _Float16* vl = &vsh[buf][0];

    // S tile 16x32: two 16x16 accumulators over 3 K-chunks (ds_load_b128)
    v8f s0 = zero8, s1 = zero8;
#pragma unroll
    for (int c = 0; c < 3; ++c) {
      const _Float16* k0 = kl + (lr)      * HDP + c * 32 + g * 16;
      const _Float16* k1 = kl + (16 + lr) * HDP + c * 32 + g * 16;
      s0 = wmma16(aq[c], ld16(k0, k0 + 8), s0);
      s1 = wmma16(aq[c], ld16(k1, k1 + 8), s1);
    }

    // Online softmax; row stats across the 16 lanes of each half-wave.
#pragma unroll
    for (int r = 0; r < 8; ++r) {
      float t = fmaxf(s0[r], s1[r]);
#pragma unroll
      for (int m = 1; m < 16; m <<= 1) t = fmaxf(t, __shfl_xor(t, m, 32));
      float mnew  = fmaxf(mx[r], t);
      float alpha = __expf(mx[r] - mnew);
      float p0 = __expf(s0[r] - mnew);
      float p1 = __expf(s1[r] - mnew);
      float rs = p0 + p1;
#pragma unroll
      for (int m = 1; m < 16; m <<= 1) rs += __shfl_xor(rs, m, 32);
      sm[r] = sm[r] * alpha + rs;
      mx[r] = mnew;
#pragma unroll
      for (int j = 0; j < 6; ++j) oacc[j][r] *= alpha;
      myp[(r + 8 * g) * 32 + lr]      = (_Float16)p0;   // P row-major 16x32
      myp[(r + 8 * g) * 32 + lr + 16] = (_Float16)p1;
    }

    // Reload P in A-matrix layout (same-wave LDS dep -> dscnt wait)
    const _Float16* pr = myp + lr * 32;
    v16h ap = ld16(pr + g * 16, pr + g * 16 + 8);

    // O += P @ V from the shared V^T tile
#pragma unroll
    for (int j = 0; j < 6; ++j) {
      const _Float16* vp = vl + (size_t)(j * 16 + lr) * 32 + g * 16;
      oacc[j] = wmma16(ap, ld16(vp, vp + 8), oacc[j]);
    }

    __syncthreads();        // done reading `buf` before it gets re-staged
    buf ^= 1;
  }

  // Writeout: oh[B,N,C] f16 with col = h*72 + d (skip hd padding)
  const int bb = bh >> 4, hh = bh & 15;
#pragma unroll
  for (int j = 0; j < 6; ++j) {
    const int d = j * 16 + lr;
    if (d < HD) {
#pragma unroll
      for (int r = 0; r < 8; ++r) {
        const int n = qt * 16 + r + 8 * g;
        oh[((size_t)bb * NTOK + n) * CIN + hh * HD + d] =
            (_Float16)(oacc[j][r] / sm[r]);
      }
    }
  }
}

// ---------------------------------------------------------------------------
// Kernel 3: out = oh @ WprojT + bproj (f32 out). Wave: 16x64 tile.
// ---------------------------------------------------------------------------
__global__ __launch_bounds__(256) void proj_kernel(
    const _Float16* __restrict__ oh, const _Float16* __restrict__ wpt,
    const float* __restrict__ bp, float* __restrict__ out) {
  const int lane = threadIdx.x & 31;
  const int wave = threadIdx.x >> 5;
  const int NB = CIN / 64;                      // 18 col blocks
  const int bm = blockIdx.x / NB, bn = blockIdx.x % NB;
  const int g = lane >> 4, lr = lane & 15;

  const _Float16* arow =
      oh + (size_t)(bm * 128 + wave * 16 + lr) * CIN;
  v8f acc[4];
  v8f z = {};
#pragma unroll
  for (int t = 0; t < 4; ++t) acc[t] = z;

  for (int kk = 0; kk < CIN; kk += 32) {
    v16h a = ld16(arow + kk + g * 16, arow + kk + g * 16 + 8);
#pragma unroll
    for (int t = 0; t < 4; ++t) {
      const _Float16* bpp =
          wpt + (size_t)(bn * 64 + t * 16 + lr) * CIN + kk + g * 16;
      acc[t] = wmma16(a, ld16(bpp, bpp + 8), acc[t]);
    }
  }
#pragma unroll
  for (int t = 0; t < 4; ++t) {
    const int col = bn * 64 + t * 16 + lr;
    const float bias = bp[col];
#pragma unroll
    for (int r = 0; r < 8; ++r)
      out[(size_t)(bm * 128 + wave * 16 + r + 8 * g) * CIN + col] =
          acc[t][r] + bias;
  }
}

// ---------------------------------------------------------------------------
// ws layout (f16): qh | kh | vt (each 6,291,456) | oh | xh (4,718,592 each)
//                 | wqt (3,981,312) | wpt (1,327,104)  => ~67 MB total
// ---------------------------------------------------------------------------
extern "C" void kernel_launch(void* const* d_in, const int* in_sizes, int n_in,
                              void* d_out, int out_size, void* d_ws,
                              size_t ws_size, hipStream_t stream) {
  (void)in_sizes; (void)n_in; (void)out_size; (void)ws_size;
  const float* x     = (const float*)d_in[0];
  const float* Wqkv  = (const float*)d_in[1];
  const float* Wproj = (const float*)d_in[2];
  const float* bproj = (const float*)d_in[3];

  const size_t QKV = (size_t)BATCH * NH * NTOK * HDP;   // 6,291,456
  const size_t XE  = (size_t)BATCH * NTOK * CIN;        // 4,718,592
  _Float16* qh  = (_Float16*)d_ws;
  _Float16* kh  = qh + QKV;
  _Float16* vt  = kh + QKV;
  _Float16* oh  = vt + QKV;
  _Float16* xh  = oh + XE;
  _Float16* wqt = xh + XE;
  _Float16* wpt = wqt + (size_t)C3 * CIN;

  const long n4 = (long)(QKV * 3 * sizeof(_Float16) / 16);
  zero_ws_kernel<<<(int)((n4 + 255) / 256), 256, 0, stream>>>((uint4*)d_ws,
                                                              n4);
  cvt_f16_kernel<<<(int)(XE / 4 / 256), 256, 0, stream>>>(x, xh, (long)XE);
  transpose_cvt_kernel<<<dim3(C3 / 32, CIN / 32), 256, 0, stream>>>(
      Wqkv, wqt, CIN, C3);
  transpose_cvt_kernel<<<dim3(CIN / 32, CIN / 32), 256, 0, stream>>>(
      Wproj, wpt, CIN, CIN);

  qkv_rope_kernel<<<32 * 54, 256, 0, stream>>>(xh, wqt, qh, kh, vt);
  attn_kernel<<<512, 256, 0, stream>>>(qh, kh, vt, oh);
  proj_kernel<<<32 * 18, 256, 0, stream>>>(oh, wpt, bproj, (float*)d_out);
}
